// FFTChainMatrix_36069135352532
// MI455X (gfx1250) — matrix operand order repaired and checked by reading, not compiled, not run on hardware
//
#include <hip/hip_runtime.h>
#include <math.h>

// ---------------------------------------------------------------------------
// Block-circulant layer on MI455X (gfx1250), fused FFT-domain path, f16 WMMA.
// fwd DFT (WMMA) -> per-freq complex GEMM (WMMA) -> inv DFT (WMMA),
// spectra in LDS (311KB of the WGP's 320KB). All fragments use vector loads;
// shared fragments are hoisted out of the job loops; W panels prefetched.
// ---------------------------------------------------------------------------

typedef __attribute__((ext_vector_type(16))) _Float16 v16h;
typedef __attribute__((ext_vector_type(8)))  _Float16 h8;
typedef __attribute__((ext_vector_type(4)))  _Float16 h4;
typedef __attribute__((ext_vector_type(8)))  float    v8f;
typedef __attribute__((ext_vector_type(4)))  float    f4;

#define TOKENS   4096      // B*S
#define NFEAT    4096
#define NBLK     64        // in-blocks == out-blocks
#define BS       64        // circulant block size
#define NF       33        // rfft bins of length-64 signal
#define NFP      48        // freq dim padded to 3 N-tiles for fwd GEMM
#define TT       16        // tokens per workgroup (one WMMA M-tile)
#define YFS      36        // YF row stride in halfs (33 + pad, 8B-aligned rows)

#define WSUM_ELEMS (NBLK*NBLK*BS)     // 262144 f32
#define WFREQ_ELEMS (NF*NBLK*NBLK)    // 135168 f16 each (Wr, Wi)
#define PI2 6.283185307179586f

static __device__ __forceinline__ v8f wmma16(v16h a, v16h b, v8f c) {
  // D = A(16x32 f16) * B(32x16 f16) + C(16x16 f32)
  return __builtin_amdgcn_wmma_f32_16x16x32_f16(false, a, false, b, (short)0, c, false, false);
}

static __device__ __forceinline__ v16h cat8(h8 a, h8 b) {
  return __builtin_shufflevector(a, b, 0,1,2,3,4,5,6,7,8,9,10,11,12,13,14,15);
}
static __device__ __forceinline__ v16h cat4(h4 a, h4 b, h4 c, h4 d) {
  h8 lo = __builtin_shufflevector(a, b, 0,1,2,3,4,5,6,7);
  h8 hi = __builtin_shufflevector(c, d, 0,1,2,3,4,5,6,7);
  return cat8(lo, hi);
}
static __device__ __forceinline__ v16h vneg(v16h a) {
  v16h r;
#pragma unroll
  for (int e = 0; e < 16; ++e) r[e] = -a[e];
  return r;
}

// --- stage 0: fold channel_weights over the 4 fft matrices (linear) --------
__global__ void k_wsum(const float* __restrict__ params,
                       const float* __restrict__ cw,
                       float* __restrict__ wsum) {
  int idx = blockIdx.x * blockDim.x + threadIdx.x;
  if (idx >= WSUM_ELEMS) return;
  wsum[idx] = cw[0]*params[idx]
            + cw[1]*params[idx +   WSUM_ELEMS]
            + cw[2]*params[idx + 2*WSUM_ELEMS]
            + cw[3]*params[idx + 3*WSUM_ELEMS];
}

// --- stage 1: rfft of each (o,i) circulant -> W[f][o][i]  (B^T layout) -----
__global__ void k_wdft(const float* __restrict__ wsum,
                       _Float16* __restrict__ Wr,
                       _Float16* __restrict__ Wi) {
  int idx = blockIdx.x * blockDim.x + threadIdx.x;   // idx = (f*64 + o)*64 + i
  if (idx >= WFREQ_ELEMS) return;
  int f = idx >> 12;
  int o = (idx >> 6) & 63;
  int i = idx & 63;
  const float* src = wsum + (o*NBLK + i)*BS;
  float sr = 0.f, si = 0.f;
  float w = PI2 * (float)f / (float)BS;
  for (int k = 0; k < BS; ++k) {
    float s, c; sincosf(w * (float)k, &s, &c);
    sr += src[k] * c;
    si -= src[k] * s;          // e^{-i theta}
  }
  Wr[idx] = (_Float16)sr;
  Wi[idx] = (_Float16)si;
}

// --- stage 2: fused fwd-DFT -> per-freq complex GEMM -> inv-DFT ------------
__global__ void __launch_bounds__(256)
k_main(const float* __restrict__ x,
       const _Float16* __restrict__ Wr,   // [f][o][i]  (transposed for b128)
       const _Float16* __restrict__ Wi,
       float* __restrict__ out) {
  extern __shared__ __align__(16) char smem_raw[];
  _Float16* FWD_C = (_Float16*)smem_raw;          // [NFP][64]  cos, transposed
  _Float16* FWD_S = FWD_C + NFP*64;               // [NFP][64]  -sin, transposed
  _Float16* INV_R = FWD_S + NFP*64;               // [64][64]   [t][f]
  _Float16* INV_I = INV_R + 64*64;
  _Float16* XFr   = INV_I + 64*64;                // [NF][TT][64]
  _Float16* XFi   = XFr + NF*TT*64;
  _Float16* YFr   = XFi + NF*TT*64;               // [TT*64][YFS]
  _Float16* YFi   = YFr + TT*64*YFS;

  const int tid  = threadIdx.x;
  const int L    = tid & 31;
  const int wv   = tid >> 5;          // 8 waves
  const int lm   = L & 15;
  const int lh   = L >> 4;
  const int tok0 = blockIdx.x * TT;

  // DFT tables, stored transposed so B-fragments are contiguous in k
  for (int n = tid; n < NFP*64; n += 256) {
    int f = n / 64, k = n % 64;
    float s = 0.f, c = 0.f;
    if (f < NF) sincosf(PI2 * (float)f * (float)k / (float)BS, &s, &c);
    FWD_C[n] = (_Float16)c;
    FWD_S[n] = (_Float16)(-s);
  }
  for (int n = tid; n < 64*64; n += 256) {
    int t = n / 64, f = n % 64;
    float vr = 0.f, vi = 0.f;
    if (f < NF) {
      float cf = (f == 0 || f == 32) ? (1.0f/(float)BS) : (2.0f/(float)BS);
      float s, c; sincosf(PI2 * (float)f * (float)t / (float)BS, &s, &c);
      vr = cf * c; vi = -cf * s;
    }
    INV_R[n] = (_Float16)vr;
    INV_I[n] = (_Float16)vi;
  }
  __syncthreads();

  // ---- Phase A: forward DFT of each input block via WMMA ----
  // D[tok][f] = sum_k x[tok][k] * C[k][f]
  {
    // Hoist the 12 DFT B-fragments (shared by all 8 i-blocks of this wave)
    v16h fbr0[3], fbr1[3], fbi0[3], fbi1[3];
#pragma unroll
    for (int t3 = 0; t3 < 3; ++t3) {
      const _Float16* pc = FWD_C + (t3*16 + lm)*64 + 16*lh;   // k = 16*lh + e
      const _Float16* ps = FWD_S + (t3*16 + lm)*64 + 16*lh;
      fbr0[t3] = cat8(*(const h8*)(pc),      *(const h8*)(pc + 8));
      fbr1[t3] = cat8(*(const h8*)(pc + 32), *(const h8*)(pc + 40));
      fbi0[t3] = cat8(*(const h8*)(ps),      *(const h8*)(ps + 8));
      fbi1[t3] = cat8(*(const h8*)(ps + 32), *(const h8*)(ps + 40));
    }
    for (int ib = wv; ib < NBLK; ib += 8) {
      const float* xp = x + (size_t)(tok0 + lm) * NFEAT + ib * BS;
      f4 c0 = *(const f4*)(xp + 16*lh);      f4 c1 = *(const f4*)(xp + 16*lh + 4);
      f4 c2 = *(const f4*)(xp + 8 + 16*lh);  f4 c3 = *(const f4*)(xp + 12 + 16*lh);
      f4 c4 = *(const f4*)(xp + 32 + 16*lh);  f4 c5 = *(const f4*)(xp + 36 + 16*lh);
      f4 c6 = *(const f4*)(xp + 40 + 16*lh);  f4 c7 = *(const f4*)(xp + 44 + 16*lh);
      v16h a0, a1;
#pragma unroll
      for (int e = 0; e < 4; ++e) {
        a0[e]    = (_Float16)c0[e];  a0[e+4]  = (_Float16)c1[e];
        a0[e+8]  = (_Float16)c2[e];  a0[e+12] = (_Float16)c3[e];
        a1[e]    = (_Float16)c4[e];  a1[e+4]  = (_Float16)c5[e];
        a1[e+8]  = (_Float16)c6[e];  a1[e+12] = (_Float16)c7[e];
      }
#pragma unroll
      for (int t3 = 0; t3 < 3; ++t3) {
        v8f dr = {}, di = {};
        dr = wmma16(a0, fbr0[t3], dr);  dr = wmma16(a1, fbr1[t3], dr);
        di = wmma16(a0, fbi0[t3], di);  di = wmma16(a1, fbi1[t3], di);
        int f = t3*16 + lm;
        if (f < NF) {
          _Float16* sr = XFr + (f*TT + 8*lh)*64 + ib;   // + r*64 per row
          _Float16* si = XFi + (f*TT + 8*lh)*64 + ib;
#pragma unroll
          for (int r = 0; r < 8; ++r) {
            sr[r*64] = (_Float16)dr[r];
            si[r*64] = (_Float16)di[r];
          }
        }
      }
    }
  }
  __syncthreads();

  // ---- Phase B: per-frequency complex GEMM  YF = XF * W[f] ----
  // Yr = Xr*Wr - Xi*Wi ; Yi = Xr*Wi + Xi*Wr   (K = 64 in-blocks)
  for (int f = wv; f < NF; f += 8) {
    // prefetch next frequency's 8KB weight panels into L2/L0
    if (f + 8 < NF) {
      __builtin_prefetch((const char*)(Wr + (size_t)(f+8)*4096) + L*256, 0, 1);
      __builtin_prefetch((const char*)(Wi + (size_t)(f+8)*4096) + L*256, 0, 1);
    }
    const _Float16* pr = XFr + (f*TT + lm)*64 + 16*lh;
    const _Float16* pi = XFi + (f*TT + lm)*64 + 16*lh;
    v16h ar0 = cat8(*(const h8*)(pr),      *(const h8*)(pr + 8));
    v16h ar1 = cat8(*(const h8*)(pr + 32), *(const h8*)(pr + 40));
    v16h ai0 = cat8(*(const h8*)(pi),      *(const h8*)(pi + 8));
    v16h ai1 = cat8(*(const h8*)(pi + 32), *(const h8*)(pi + 40));
    v16h an0 = vneg(ai0), an1 = vneg(ai1);

#pragma unroll
    for (int ot = 0; ot < 4; ++ot) {
      const int col = ot*16 + lm;
      const _Float16* qr = Wr + ((size_t)f*64 + col)*64 + 16*lh;
      const _Float16* qi = Wi + ((size_t)f*64 + col)*64 + 16*lh;
      v16h br0 = cat8(*(const h8*)(qr),      *(const h8*)(qr + 8));
      v16h br1 = cat8(*(const h8*)(qr + 32), *(const h8*)(qr + 40));
      v16h bi0 = cat8(*(const h8*)(qi),      *(const h8*)(qi + 8));
      v16h bi1 = cat8(*(const h8*)(qi + 32), *(const h8*)(qi + 40));

      v8f dr = {}, di = {};
      dr = wmma16(ar0, br0, dr);  dr = wmma16(ar1, br1, dr);
      dr = wmma16(an0, bi0, dr);  dr = wmma16(an1, bi1, dr);
      di = wmma16(ar0, bi0, di);  di = wmma16(ar1, bi1, di);
      di = wmma16(ai0, br0, di);  di = wmma16(ai1, br1, di);

      _Float16* yr = YFr + ((8*lh)*64 + col)*YFS + f;   // + r*64*YFS per row
      _Float16* yi = YFi + ((8*lh)*64 + col)*YFS + f;
#pragma unroll
      for (int r = 0; r < 8; ++r) {
        yr[r*64*YFS] = (_Float16)dr[r];
        yi[r*64*YFS] = (_Float16)di[r];
      }
    }
  }
  __syncthreads();

  // ---- Phase C: inverse DFT GEMM over frequencies -> global output ----
  // y[(tok,o)][t] = sum_f YFr*IR[f][t] + YFi*II[f][t]
  {
    // Hoist the 12 inverse-DFT B-fragments (shared by all 8 M-tiles)
    v16h brr0[4], brr1[4], bri0[4];
#pragma unroll
    for (int tt = 0; tt < 4; ++tt) {
      const _Float16* qr = INV_R + (tt*16 + lm)*64 + 16*lh;
      const _Float16* qi = INV_I + (tt*16 + lm)*64 + 16*lh;
      brr0[tt] = cat8(*(const h8*)(qr),      *(const h8*)(qr + 8));
      brr1[tt] = cat8(*(const h8*)(qr + 32), *(const h8*)(qr + 40));
      bri0[tt] = cat8(*(const h8*)(qi),      *(const h8*)(qi + 8));
    }
    for (int mt = wv; mt < 64; mt += 8) {
      const _Float16* pr = YFr + (mt*16 + lm)*YFS + 16*lh;  // 8B-aligned chunks
      const _Float16* pi = YFi + (mt*16 + lm)*YFS + 16*lh;
      v16h ar0 = cat4(*(const h4*)(pr),      *(const h4*)(pr + 4),
                      *(const h4*)(pr + 8), *(const h4*)(pr + 12));
      v16h ai0 = cat4(*(const h4*)(pi),      *(const h4*)(pi + 4),
                      *(const h4*)(pi + 8), *(const h4*)(pi + 12));
      // k-step 1: only bin f=32 is live (element 0 of lanes 0..15)
      v16h ar1 = {};
      if (lh == 0) ar1[0] = YFr[(mt*16 + lm)*YFS + 32];

#pragma unroll
      for (int tt = 0; tt < 4; ++tt) {
        v8f d = {};
        d = wmma16(ar0, brr0[tt], d);
        d = wmma16(ar1, brr1[tt], d);
        d = wmma16(ai0, bri0[tt], d);
        float* op = out + (size_t)(tok0 + (mt >> 2))*NFEAT
                  + ((mt & 3)*16 + 8*lh)*BS + tt*16 + lm;   // row = mt*16+..
#pragma unroll
        for (int r = 0; r < 8; ++r) {
          int row  = mt*16 + r + 8*lh;
          int tokl = row >> 6;
          int o    = row & 63;
          out[(size_t)(tok0 + tokl)*NFEAT + o*BS + tt*16 + lm] = d[r];
        }
        (void)op;
      }
    }
  }
}

// ---------------------------------------------------------------------------
extern "C" void kernel_launch(void* const* d_in, const int* in_sizes, int n_in,
                              void* d_out, int out_size, void* d_ws, size_t ws_size,
                              hipStream_t stream) {
  const float* x      = (const float*)d_in[0];   // (2,2048,4096) f32
  const float* params = (const float*)d_in[1];   // (4,64,64,64) f32
  const float* cw     = (const float*)d_in[2];   // (4,) f32

  float*    wsum = (float*)d_ws;                                   // 1 MiB
  _Float16* Wr   = (_Float16*)((char*)d_ws + WSUM_ELEMS*sizeof(float));
  _Float16* Wi   = Wr + WFREQ_ELEMS;

  k_wsum<<<(WSUM_ELEMS + 255)/256, 256, 0, stream>>>(params, cw, wsum);
  k_wdft<<<(WFREQ_ELEMS + 255)/256, 256, 0, stream>>>(wsum, Wr, Wi);

  // LDS: tables(28KB) + XF(132KB) + YF(144KB) = 311296 B  (< 320KB/WGP)
  const size_t smem = (size_t)(NFP*64*2 + 64*64*2 + NF*TT*64*2 + TT*64*YFS*2)
                      * sizeof(_Float16);
  k_main<<<TOKENS/TT, 256, smem, stream>>>(x, Wr, Wi, (float*)d_out);
}
